// Attention_6622839570608
// MI455X (gfx1250) — matrix operand order repaired and checked by reading, not compiled, not run on hardware
//
#include <hip/hip_runtime.h>

// ---------------------------------------------------------------------------
// Problem constants (from reference setup_inputs)
// ---------------------------------------------------------------------------
#define BB   4
#define NN   4096        // tokens
#define CC   512         // channels
#define NH   8           // heads
#define HD   64          // head dim
#define MM   1024        // reduced tokens = (64/2)*(64/2)
#define SCALE 0.125f     // hd^-0.5

// ---------------------------------------------------------------------------
// WMMA types (CDNA5 / gfx1250, wave32)
// ---------------------------------------------------------------------------
typedef __attribute__((ext_vector_type(16))) __bf16 v16bf;
typedef __attribute__((ext_vector_type(8)))  float  v8f;

union ABu {
    v16bf          v;
    unsigned int   u[8];
    unsigned short s[16];
};

__device__ __forceinline__ unsigned short f2bf(float f) {
    unsigned int u = __float_as_uint(f);
    u += 0x7FFFu + ((u >> 16) & 1u);   // round-to-nearest-even
    return (unsigned short)(u >> 16);
}

__device__ __forceinline__ v8f wmma_bf16(const ABu& a, const ABu& b, v8f c) {
    // D = A(16x32 bf16) * B(32x16 bf16) + C(16x16 f32)
    return __builtin_amdgcn_wmma_f32_16x16x32_bf16(
        /*neg_a=*/false, a.v, /*neg_b=*/false, b.v,
        /*c_mod=*/(short)0, c, /*reuse_a=*/false, /*reuse_b=*/false);
}

// ---------------------------------------------------------------------------
// gfx1250 async global->LDS copy (ASYNCcnt-tracked DMA, bypasses VGPRs)
// lds_off = low 32 bits of the flat LDS address (aperture decode truncates).
// ---------------------------------------------------------------------------
__device__ __forceinline__ void async_ld16(unsigned lds_off, const void* gptr) {
    asm volatile("global_load_async_to_lds_b128 %0, %1, off"
                 :: "v"(lds_off),
                    "v"((unsigned long long)(uintptr_t)gptr)
                 : "memory");
}
__device__ __forceinline__ void wait_async0() {
    asm volatile("s_wait_asynccnt 0x0" ::: "memory");
}

// ---------------------------------------------------------------------------
// fp32 -> bf16 elementwise convert
// ---------------------------------------------------------------------------
__global__ void cvt_bf16_kernel(const float* __restrict__ in,
                                unsigned short* __restrict__ out, int n) {
    int i = blockIdx.x * 256 + threadIdx.x;
    if (i < n) out[i] = f2bf(in[i]);
}

// ---------------------------------------------------------------------------
// Depthwise 2x2 stride-2 conv + bias + LayerNorm -> bf16 [B*M, C]
// One wave per output token; 16 channels per lane.
// ---------------------------------------------------------------------------
__global__ __launch_bounds__(128)
void conv_ln_kernel(const float* __restrict__ x,      // [B, N, C]
                    const float* __restrict__ sr_w,   // [C,1,2,2]
                    const float* __restrict__ sr_b,   // [C]
                    const float* __restrict__ ln_g,   // [C]
                    const float* __restrict__ ln_b,   // [C]
                    unsigned short* __restrict__ xr)  // [B*M, C] bf16
{
    int wave = threadIdx.x >> 5;
    int lane = threadIdx.x & 31;
    int idx  = blockIdx.x * 4 + wave;       // 0 .. B*M-1
    int b = idx >> 10;
    int m = idx & (MM - 1);
    int i = m >> 5, j = m & 31;             // output pixel in 32x32
    const float* xb = x + (size_t)b * NN * CC;
    int n00 = (2 * i) * 64 + 2 * j;         // H=W=64 row-major

    float y[16];
    float s1 = 0.f, s2 = 0.f;
#pragma unroll
    for (int t = 0; t < 16; ++t) {
        int c = t * 32 + lane;
        float w00 = sr_w[c * 4 + 0], w01 = sr_w[c * 4 + 1];
        float w10 = sr_w[c * 4 + 2], w11 = sr_w[c * 4 + 3];
        float v = xb[(size_t)(n00     ) * CC + c] * w00
                + xb[(size_t)(n00 +  1) * CC + c] * w01
                + xb[(size_t)(n00 + 64) * CC + c] * w10
                + xb[(size_t)(n00 + 65) * CC + c] * w11
                + sr_b[c];
        y[t] = v; s1 += v; s2 += v * v;
    }
#pragma unroll
    for (int off = 16; off > 0; off >>= 1) {
        s1 += __shfl_xor(s1, off);
        s2 += __shfl_xor(s2, off);
    }
    float mean = s1 * (1.0f / CC);
    float var  = s2 * (1.0f / CC) - mean * mean;
    float inv  = rsqrtf(var + 1e-5f);
#pragma unroll
    for (int t = 0; t < 16; ++t) {
        int c = t * 32 + lane;
        float v = (y[t] - mean) * inv * ln_g[c] + ln_b[c];
        xr[(size_t)idx * CC + c] = f2bf(v);
    }
}

// ---------------------------------------------------------------------------
// Generic bf16 WMMA GEMM:  out = A[rows,K] @ Bt[cols,K]^T
// Block = 128 threads (4 waves) computing a 64x64 output tile.
// Tile staging uses gfx1250 async global->LDS DMA, double-buffered so the
// next tile streams in while the current tile's WMMAs execute.
// Store modes:
//   0 : fp32 + bias, row-major ldc=CC (final projection into d_out)
//   2 : q head-split  -> q_bf [B, NH, NN, HD]
//   3 : kv split      -> k_bf [B, NH, MM, HD]  and vT_bf [B, NH, HD, MM]
// ---------------------------------------------------------------------------
__global__ __launch_bounds__(128)
void gemm_bf16_wmma(const unsigned short* __restrict__ A,
                    const unsigned short* __restrict__ Bt,
                    int K, int mode,
                    float* __restrict__ outF,
                    unsigned short* __restrict__ out0,
                    unsigned short* __restrict__ out1,
                    const float* __restrict__ bias)
{
    __shared__ alignas(16) unsigned short Asm[2][64 * 32];
    __shared__ alignas(16) unsigned short Bsm[2][64 * 32];

    int tid  = threadIdx.x;
    int wave = tid >> 5;
    int lane = tid & 31;
    int n    = lane & 15;
    int hi   = lane >> 4;
    int rowBase = blockIdx.y * 64;
    int colBase = blockIdx.x * 64;
    int m_local = (wave << 4) + n;

    // staging assignment: thread covers 32B of one tile row in each of A/B
    int r  = tid >> 1;
    int cE = (tid & 1) * 16;          // element offset within 32-elem row

    auto issueTile = [&](int buf, int k0) {
        const unsigned short* ga = A  + (size_t)(rowBase + r) * K + k0 + cE;
        const unsigned short* gb = Bt + (size_t)(colBase + r) * K + k0 + cE;
        unsigned la = (unsigned)(uintptr_t)&Asm[buf][r * 32 + cE];
        unsigned lb = (unsigned)(uintptr_t)&Bsm[buf][r * 32 + cE];
        async_ld16(la,      ga);
        async_ld16(la + 16, (const char*)ga + 16);
        async_ld16(lb,      gb);
        async_ld16(lb + 16, (const char*)gb + 16);
    };

    v8f acc[4] = {};
    int nIter = K / 32;
    issueTile(0, 0);

    for (int it = 0; it < nIter; ++it) {
        wait_async0();                 // this tile's DMA complete (this wave)
        __syncthreads();               // all waves' DMA published
        if (it + 1 < nIter) issueTile((it + 1) & 1, (it + 1) * 32);

        int buf = it & 1;
        // A fragment (16x32): lane holds row m_local, K per ISA layout
        ABu af;
        const unsigned int* Arow = (const unsigned int*)&Asm[buf][m_local * 32];
#pragma unroll
        for (int v = 0; v < 4; ++v) af.u[v]     = Arow[hi * 4 + v];       // K = hi*8+2v
#pragma unroll
        for (int v = 0; v < 4; ++v) af.u[4 + v] = Arow[8 + hi * 4 + v];   // K = 16+hi*8+2v

#pragma unroll
        for (int ct = 0; ct < 4; ++ct) {
            ABu bf;
            const unsigned int* Brow =
                (const unsigned int*)&Bsm[buf][(ct * 16 + n) * 32];
#pragma unroll
            for (int v = 0; v < 8; ++v) bf.u[v] = Brow[(v >> 2) * 8 + hi * 4 + (v & 3)];       // K = (v>>2)*16+hi*8+2(v&3)
            acc[ct] = wmma_bf16(af, bf, acc[ct]);
        }
        __syncthreads();               // reads done before buffer reuse
    }

    // store: D element (M = v + 8*hi, N = n) per tile
    if (mode == 0) {
#pragma unroll
        for (int ct = 0; ct < 4; ++ct)
#pragma unroll
            for (int v = 0; v < 8; ++v) {
                int R = rowBase + (wave << 4) + v + 8 * hi;
                int O = colBase + ct * 16 + n;
                outF[(size_t)R * CC + O] = acc[ct][v] + bias[O];
            }
    } else if (mode == 2) {
#pragma unroll
        for (int ct = 0; ct < 4; ++ct)
#pragma unroll
            for (int v = 0; v < 8; ++v) {
                int R = rowBase + (wave << 4) + v + 8 * hi;
                int O = colBase + ct * 16 + n;
                int b = R >> 12, nn = R & (NN - 1);
                int h = O >> 6,  d  = O & (HD - 1);
                out0[(((size_t)(b * NH + h) * NN) + nn) * HD + d] = f2bf(acc[ct][v]);
            }
    } else { // mode 3: kv split
#pragma unroll
        for (int ct = 0; ct < 4; ++ct)
#pragma unroll
            for (int v = 0; v < 8; ++v) {
                int R = rowBase + (wave << 4) + v + 8 * hi;
                int O = colBase + ct * 16 + n;
                int b = R >> 10, mm = R & (MM - 1);
                if (O < CC) {
                    int h = O >> 6, d = O & (HD - 1);
                    out0[(((size_t)(b * NH + h) * MM) + mm) * HD + d] = f2bf(acc[ct][v]);
                } else {
                    int O2 = O - CC;
                    int h = O2 >> 6, d = O2 & (HD - 1);
                    out1[(((size_t)(b * NH + h) * HD) + d) * MM + mm] = f2bf(acc[ct][v]);
                }
            }
    }
}

// ---------------------------------------------------------------------------
// Fused attention: one wave per (b, h, 16-query-row tile).
// scores = softmax(q k^T * scale) over M=1024, then out = P @ v.
// ---------------------------------------------------------------------------
__global__ __launch_bounds__(32)
void attn_kernel(const unsigned short* __restrict__ q,    // [B, NH, NN, HD] bf16
                 const unsigned short* __restrict__ k,    // [B, NH, MM, HD] bf16
                 const unsigned short* __restrict__ vT,   // [B, NH, HD, MM] bf16
                 unsigned short* __restrict__ ao)         // [B*NN, CC] bf16
{
    __shared__ float sc[16 * MM];        // 64 KB score buffer

    int lane = threadIdx.x;
    int n    = lane & 15;
    int hi   = lane >> 4;
    int qt   = blockIdx.x;               // 0..255 (query tile)
    int bh   = blockIdx.y;               // 0..31  (b*NH + h)

    const unsigned short* qB = q  + ((size_t)bh * NN + qt * 16) * HD;
    const unsigned short* kB = k  + (size_t)bh * MM * HD;
    const unsigned short* vB = vT + (size_t)bh * HD * MM;

    // Q fragments: two K-chunks of 32 covering hd=64
    ABu aq[2];
#pragma unroll
    for (int kc = 0; kc < 2; ++kc) {
        const unsigned int* row = (const unsigned int*)(qB + (size_t)n * HD + kc * 32);
#pragma unroll
        for (int v = 0; v < 4; ++v) aq[kc].u[v]     = row[hi * 4 + v];
#pragma unroll
        for (int v = 0; v < 4; ++v) aq[kc].u[4 + v] = row[8 + hi * 4 + v];
    }

    // scores -> LDS
    for (int mt = 0; mt < MM / 16; ++mt) {
        if (mt + 8 < MM / 16)            // stream K rows ahead of use
            __builtin_prefetch(kB + (size_t)((mt + 8) * 16 + n) * HD, 0, 1);
        v8f d = {};
#pragma unroll
        for (int kc = 0; kc < 2; ++kc) {
            ABu bf;
            const unsigned int* row =
                (const unsigned int*)(kB + (size_t)(mt * 16 + n) * HD + kc * 32);
#pragma unroll
            for (int v = 0; v < 8; ++v) bf.u[v] = row[(v >> 2) * 8 + hi * 4 + (v & 3)];
            d = wmma_bf16(aq[kc], bf, d);
        }
#pragma unroll
        for (int v = 0; v < 8; ++v)
            sc[(v + 8 * hi) * MM + mt * 16 + n] = d[v] * SCALE;
    }
    __syncthreads();

    // softmax over M: row = lane&15, half hi handles 512 cols; fp32 stats
    float* rowp = sc + n * MM + hi * 512;
    float mx = -3.0e38f;
    for (int i = 0; i < 512; ++i) mx = fmaxf(mx, rowp[i]);
    mx = fmaxf(mx, __shfl_xor(mx, 16));
    float sum = 0.f;
    for (int i = 0; i < 512; ++i) {
        float e = __expf(rowp[i] - mx);
        rowp[i] = e;
        sum += e;
    }
    sum += __shfl_xor(sum, 16);
    float rs = 1.0f / sum;               // lanes L and L^16 agree; row = L&15
    __syncthreads();

    // out = P @ v  (K = M = 1024, 32 chunks; 4 output d-tiles)
    v8f o[4] = {};
    for (int mt = 0; mt < MM / 32; ++mt) {
        if (mt + 4 < MM / 32)            // stream V rows ahead of use
            __builtin_prefetch(vB + (size_t)n * MM + (mt + 4) * 32, 0, 1);
        ABu ap;
        const float* prow = sc + n * MM + mt * 32;
#pragma unroll
        for (int v = 0; v < 4; ++v) {
            int kk = hi * 8 + 2 * v;
            ap.s[2 * v]     = f2bf(prow[kk]);
            ap.s[2 * v + 1] = f2bf(prow[kk + 1]);
        }
#pragma unroll
        for (int v = 0; v < 4; ++v) {
            int kk = 16 + hi * 8 + 2 * v;
            ap.s[8 + 2 * v]     = f2bf(prow[kk]);
            ap.s[8 + 2 * v + 1] = f2bf(prow[kk + 1]);
        }
#pragma unroll
        for (int dt = 0; dt < 4; ++dt) {
            ABu bf;
            const unsigned int* row =
                (const unsigned int*)(vB + (size_t)(dt * 16 + n) * MM + mt * 32);
#pragma unroll
            for (int v = 0; v < 8; ++v) bf.u[v] = row[(v >> 2) * 8 + hi * 4 + (v & 3)];
            o[dt] = wmma_bf16(ap, bf, o[dt]);
        }
    }

    // epilogue: scale by 1/rowsum, store bf16 into [B*NN, CC]
    int b = bh >> 3, h = bh & 7;
#pragma unroll
    for (int dt = 0; dt < 4; ++dt)
#pragma unroll
        for (int v = 0; v < 8; ++v) {
            int M   = v + 8 * hi;
            float r = __shfl(rs, M);
            float val = o[dt][v] * r;
            int nrow = qt * 16 + M;
            int c    = h * HD + dt * 16 + n;
            ao[((size_t)b * NN + nrow) * CC + c] = f2bf(val);
        }
}

// ---------------------------------------------------------------------------
// Launch
// ---------------------------------------------------------------------------
extern "C" void kernel_launch(void* const* d_in, const int* in_sizes, int n_in,
                              void* d_out, int out_size, void* d_ws, size_t ws_size,
                              hipStream_t stream) {
    const float* x      = (const float*)d_in[0];
    const float* q_w    = (const float*)d_in[1];
    const float* kv_w   = (const float*)d_in[2];
    const float* proj_w = (const float*)d_in[3];
    const float* proj_b = (const float*)d_in[4];
    const float* sr_w   = (const float*)d_in[5];
    const float* sr_b   = (const float*)d_in[6];
    const float* ln_g   = (const float*)d_in[7];
    const float* ln_b   = (const float*)d_in[8];
    float* out = (float*)d_out;

    char* ws = (char*)d_ws;
    size_t off = 0;
    auto alloc = [&](size_t bytes) -> void* {
        void* p = ws + off;
        off += (bytes + 255) & ~(size_t)255;
        return p;
    };
    unsigned short* x_bf   = (unsigned short*)alloc((size_t)BB * NN * CC * 2);
    unsigned short* q_bf   = (unsigned short*)alloc((size_t)BB * NN * CC * 2);
    unsigned short* xr_bf  = (unsigned short*)alloc((size_t)BB * MM * CC * 2);
    unsigned short* k_bf   = (unsigned short*)alloc((size_t)BB * NH * MM * HD * 2);
    unsigned short* vT_bf  = (unsigned short*)alloc((size_t)BB * NH * HD * MM * 2);
    unsigned short* ao_bf  = (unsigned short*)alloc((size_t)BB * NN * CC * 2);
    unsigned short* qw_bf  = (unsigned short*)alloc((size_t)CC * CC * 2);
    unsigned short* kvw_bf = (unsigned short*)alloc((size_t)2 * CC * CC * 2);
    unsigned short* pw_bf  = (unsigned short*)alloc((size_t)CC * CC * 2);

    // 1) fp32 -> bf16 converts
    {
        int nx = BB * NN * CC;
        cvt_bf16_kernel<<<(nx + 255) / 256, 256, 0, stream>>>(x, x_bf, nx);
        int nw = CC * CC;
        cvt_bf16_kernel<<<(nw + 255) / 256, 256, 0, stream>>>(q_w, qw_bf, nw);
        int nkv = 2 * CC * CC;
        cvt_bf16_kernel<<<(nkv + 255) / 256, 256, 0, stream>>>(kv_w, kvw_bf, nkv);
        cvt_bf16_kernel<<<(nw + 255) / 256, 256, 0, stream>>>(proj_w, pw_bf, nw);
    }

    // 2) depthwise conv + LN
    conv_ln_kernel<<<(BB * MM) / 4, 128, 0, stream>>>(x, sr_w, sr_b, ln_g, ln_b, xr_bf);

    // 3) q = x @ q_w^T  -> [B, NH, NN, HD]
    {
        dim3 grid(CC / 64, (BB * NN) / 64);
        gemm_bf16_wmma<<<grid, 128, 0, stream>>>(x_bf, qw_bf, CC, 2,
                                                 nullptr, q_bf, nullptr, nullptr);
    }

    // 4) kv = xr @ kv_w^T -> k [B,NH,MM,HD], vT [B,NH,HD,MM]
    {
        dim3 grid((2 * CC) / 64, (BB * MM) / 64);
        gemm_bf16_wmma<<<grid, 128, 0, stream>>>(xr_bf, kvw_bf, CC, 3,
                                                 nullptr, k_bf, vT_bf, nullptr);
    }

    // 5) fused attention -> ao_bf [B*NN, CC]
    {
        dim3 grid(NN / 16, BB * NH);
        attn_kernel<<<grid, 32, 0, stream>>>(q_bf, k_bf, vT_bf, ao_bf);
    }

    // 6) out = ao @ proj_w^T + proj_b  (fp32)
    {
        dim3 grid(CC / 64, (BB * NN) / 64);
        gemm_bf16_wmma<<<grid, 128, 0, stream>>>(ao_bf, pw_bf, CC, 0,
                                                 out, nullptr, nullptr, proj_b);
    }
    (void)in_sizes; (void)n_in; (void)out_size; (void)ws_size;
}
